// sa_layer_33243046871285
// MI455X (gfx1250) — compile-verified, matrix operand order checked by reading, not run on hardware
//
#include <hip/hip_runtime.h>

#define EPS_F 1e-8f

typedef __attribute__((ext_vector_type(16))) _Float16 v16h;
typedef __attribute__((ext_vector_type(8)))  float    v8f;
typedef __attribute__((ext_vector_type(4)))  float    v4f;   // native vector for NT builtins

union V16H { _Float16 h[16]; v16h v; };
union V8F  { float    f[8];  v8f  v; };

__device__ __forceinline__ unsigned pcg_hash(unsigned s) {
  s ^= s >> 17; s *= 0xed5ad4bbu;
  s ^= s >> 11; s *= 0xac4c1b51u;
  s ^= s >> 15; s *= 0x31848babu;
  s ^= s >> 14;
  return s;
}

// LDS weight cache layout (122 floats):
//   [0..8]    w1_1 (3x3)
//   [9..80]   w5   (4x2x3x3 = 72)
//   [81..84]  b5   (4)
//   [85..120] w6   (1x4x3x3 = 36)
//   [121]     b6   (1)
#define WT_W11 0
#define WT_W5  9
#define WT_B5  81
#define WT_W6  85
#define WT_B6  121

// One workgroup (8 wave32) per batch image. Fully fused:
//   pass1 channel-max -> LDS, tiny gate math (+WMMA convs), pass2 apply gate.
__global__ __launch_bounds__(256)
void sa_layer_fused(const float* __restrict__ x,
                    const float* __restrict__ w1_1,
                    const float* __restrict__ w5,
                    const float* __restrict__ b5,
                    const float* __restrict__ w6,
                    const float* __restrict__ b6,
                    float* __restrict__ out)
{
  const int b   = blockIdx.x;     // batch index
  const int tid = threadIdx.x;    // 256 threads
  const int pg  = tid & 15;       // pixel group: pixels 4*pg .. 4*pg+3
  const int cs  = tid >> 4;       // channel slice 0..15

  __shared__ float red[16][64];   // partial channel maxes
  __shared__ float m_s[64];       // channel max image (8x8)
  __shared__ float pre_s[16];     // pooled 4x4 image
  __shared__ float c1_s[16];      // after 1->1 conv
  __shared__ float up_s[64];      // bilinear upsampled 8x8
  __shared__ float h_s[4][64];    // hidden 4ch 8x8
  __shared__ float gate_s[64];    // sigmoid gate 8x8
  __shared__ float wt[128];       // cached weights/biases

  const float* xb = x + (size_t)b * 65536u;   // [1024, 64] slice

  // ---------- Phase 0: weights -> LDS (covered by phase-1 barrier) ----------
  if (tid < 9)        wt[tid] = w1_1[tid];
  else if (tid < 81)  wt[tid] = w5[tid - 9];
  else if (tid < 85)  wt[tid] = b5[tid - 81];
  else if (tid < 121) wt[tid] = w6[tid - 85];
  else if (tid == 121) wt[tid] = b6[0];

  // ---------- Phase 1: channel max over C=1024 (coalesced b128 stream, RT) ----------
  v4f pm;
  pm.x = pm.y = pm.z = pm.w = -__builtin_inff();
  for (int c = cs; c < 1024; c += 16) {
    const v4f v = *(const v4f*)(xb + c * 64 + pg * 4);
    pm.x = fmaxf(pm.x, v.x);
    pm.y = fmaxf(pm.y, v.y);
    pm.z = fmaxf(pm.z, v.z);
    pm.w = fmaxf(pm.w, v.w);
  }
  red[cs][pg * 4 + 0] = pm.x;
  red[cs][pg * 4 + 1] = pm.y;
  red[cs][pg * 4 + 2] = pm.z;
  red[cs][pg * 4 + 3] = pm.w;
  __syncthreads();
  if (tid < 64) {
    float v = red[0][tid];
    #pragma unroll
    for (int s = 1; s < 16; ++s) v = fmaxf(v, red[s][tid]);
    m_s[tid] = v;
  }
  __syncthreads();

  // ---------- Phase 2: 2x2 windows, stochastic/max/avg pooling mix ----------
  if (tid < 16) {
    const int r = tid >> 2, cw = tid & 3;
    float wv[4];
    wv[0] = m_s[(2 * r    ) * 8 + 2 * cw    ];
    wv[1] = m_s[(2 * r    ) * 8 + 2 * cw + 1];
    wv[2] = m_s[(2 * r + 1) * 8 + 2 * cw    ];
    wv[3] = m_s[(2 * r + 1) * 8 + 2 * cw + 1];
    float wc[4];
    #pragma unroll
    for (int k = 0; k < 4; ++k) {
      const float v = wv[k];
      // NaN (!(v>=0)), negative, -inf (!(v>=0)), +inf (==inf) -> EPS
      wc[k] = (!(v >= 0.0f) || (v == __builtin_inff())) ? EPS_F : v;
    }
    // categorical sample over softmax(wc) with deterministic per-window hash
    const float mx = fmaxf(fmaxf(wc[0], wc[1]), fmaxf(wc[2], wc[3]));
    float e[4], sum = 0.f;
    #pragma unroll
    for (int k = 0; k < 4; ++k) { e[k] = __expf(wc[k] - mx); sum += e[k]; }
    const float u = (float)(pcg_hash(0x9e3779b9u ^ (unsigned)(b * 16 + tid)) >> 8)
                    * (1.0f / 16777216.0f) * sum;
    int idx = 3; float acc = 0.f;
    #pragma unroll
    for (int k = 0; k < 3; ++k) { acc += e[k]; if (u < acc && idx == 3) idx = k; }
    const float mo2 = wc[idx];
    const float mo3 = fmaxf(fmaxf(wv[0], wv[1]), fmaxf(wv[2], wv[3]));
    const float ao4 = 0.25f * (wv[0] + wv[1] + wv[2] + wv[3]);
    pre_s[tid] = 0.1f * mo2 + 0.6f * mo3 + 0.3f * ao4;
  }
  __syncthreads();

  // ---------- Phase 3: 1->1 3x3 SAME conv on 4x4 (no bias) ----------
  if (tid < 16) {
    const int r = tid >> 2, cw = tid & 3;
    float acc = 0.f;
    #pragma unroll
    for (int dy = 0; dy < 3; ++dy)
      #pragma unroll
      for (int dx = 0; dx < 3; ++dx) {
        const int yy = r + dy - 1, xx = cw + dx - 1;
        if (yy >= 0 && yy < 4 && xx >= 0 && xx < 4)
          acc += pre_s[yy * 4 + xx] * wt[WT_W11 + dy * 3 + dx];
      }
    c1_s[tid] = acc;
  }
  __syncthreads();

  // ---------- Phase 4: bilinear 4x4 -> 8x8 (align_corners=false) ----------
  if (tid < 64) {
    const int i = tid >> 3, j = tid & 7;
    const float sy = 0.5f * (float)i - 0.25f;
    const float sx = 0.5f * (float)j - 0.25f;
    const int y0 = (int)floorf(sy); const float ty = sy - (float)y0;
    const int x0 = (int)floorf(sx); const float tx = sx - (float)x0;
    const int y0c = min(max(y0, 0), 3), y1c = min(max(y0 + 1, 0), 3);
    const int x0c = min(max(x0, 0), 3), x1c = min(max(x0 + 1, 0), 3);
    const float v00 = c1_s[y0c * 4 + x0c], v01 = c1_s[y0c * 4 + x1c];
    const float v10 = c1_s[y1c * 4 + x0c], v11 = c1_s[y1c * 4 + x1c];
    up_s[tid] = (1.f - ty) * ((1.f - tx) * v00 + tx * v01)
              +        ty  * ((1.f - tx) * v10 + tx * v11);
  }
  __syncthreads();

  // ---------- Phase 5: gate convs as WMMA GEMMs (waves 0..3, EXEC all-ones) ----------
  const int wave = tid >> 5;
  const int lane = tid & 31;
  if (wave < 4) {
    const int half = lane >> 4;       // A/B layout half
    const int nlo  = lane & 15;       // A row within half-group / B-C column N
    const int tile = wave;            // pixel tile: pixels tile*16 .. tile*16+15
    const int p    = tile * 16 + nlo; // pixel for this lane's A row
    const int py   = p >> 3, px = p & 7;

    // ----- conv2: im2col GEMM, M=16 pixels, K=18 (2ch x 3x3, pad->32), N=4 och -----
    V16H A, Bm;
    #pragma unroll
    for (int e = 0; e < 16; ++e) {
      const int i   = e >> 1, bit = e & 1;
      const int K   = (i < 4) ? (8 * half + 2 * i + bit)
                              : (16 + 8 * half + 2 * (i - 4) + bit);
      float av = 0.f, bv = 0.f;
      if (K < 18) {
        const int ich = K / 9, rem = K % 9, dy = rem / 3, dx = rem % 3;
        const int yy = py + dy - 1, xx = px + dx - 1;
        if (yy >= 0 && yy < 8 && xx >= 0 && xx < 8)
          av = (ich == 0) ? m_s[yy * 8 + xx] : up_s[yy * 8 + xx];
        if (nlo < 4) bv = wt[WT_W5 + nlo * 18 + K];   // OIHW [4][2][3][3] flat
      }
      A.h[e]  = (_Float16)av;
      Bm.h[e] = (_Float16)bv;
    }
    V8F C2 = {};
    C2.v = __builtin_amdgcn_wmma_f32_16x16x32_f16(
        false, A.v, false, Bm.v, (short)0, C2.v, false, false);
    if (nlo < 4) {                       // column N = och
      const float bias = wt[WT_B5 + nlo];
      #pragma unroll
      for (int r = 0; r < 8; ++r) {      // C row = half*8 + r
        const int pp = tile * 16 + half * 8 + r;
        h_s[nlo][pp] = fmaxf(C2.f[r] + bias, 0.f);
      }
    }
  }
  __syncthreads();

  if (wave < 4) {
    const int half = lane >> 4;
    const int nlo  = lane & 15;
    const int tile = wave;
    const int p    = tile * 16 + nlo;
    const int py   = p >> 3, px = p & 7;

    // ----- conv3: K=36 (4ch x 3x3) split into two accumulating K=32 WMMAs, N=1 -----
    V16H A0, B0, A1, B1;
    #pragma unroll
    for (int e = 0; e < 16; ++e) {
      const int i = e >> 1, bit = e & 1;
      const int K = (i < 4) ? (8 * half + 2 * i + bit)
                            : (16 + 8 * half + 2 * (i - 4) + bit);
      // chunk 0: k = K (0..31)
      float a0 = 0.f, b0 = 0.f;
      {
        const int k = K;
        const int och = k / 9, rem = k % 9, dy = rem / 3, dx = rem % 3;
        const int yy = py + dy - 1, xx = px + dx - 1;
        if (yy >= 0 && yy < 8 && xx >= 0 && xx < 8) a0 = h_s[och][yy * 8 + xx];
        if (nlo == 0) b0 = wt[WT_W6 + k];
      }
      // chunk 1: k = 32 + K, valid only for K < 4
      float a1 = 0.f, b1 = 0.f;
      if (K < 4) {
        const int k = 32 + K;
        const int och = k / 9, rem = k % 9, dy = rem / 3, dx = rem % 3;
        const int yy = py + dy - 1, xx = px + dx - 1;
        if (yy >= 0 && yy < 8 && xx >= 0 && xx < 8) a1 = h_s[och][yy * 8 + xx];
        if (nlo == 0) b1 = wt[WT_W6 + k];
      }
      A0.h[e] = (_Float16)a0;  B0.h[e] = (_Float16)b0;
      A1.h[e] = (_Float16)a1;  B1.h[e] = (_Float16)b1;
    }
    V8F Cg = {};
    Cg.v = __builtin_amdgcn_wmma_f32_16x16x32_f16(
        false, A0.v, false, B0.v, (short)0, Cg.v, false, false);
    Cg.v = __builtin_amdgcn_wmma_f32_16x16x32_f16(
        false, A1.v, false, B1.v, (short)0, Cg.v, false, false);
    if (nlo == 0) {                      // only column N=0 is the real output
      const float bias = wt[WT_B6];
      #pragma unroll
      for (int r = 0; r < 8; ++r) {
        const int pp = tile * 16 + half * 8 + r;
        gate_s[pp] = 1.f / (1.f + __expf(-(Cg.f[r] + bias)));
      }
    }
  }
  __syncthreads();

  // ---------- Phase 6: out = relu(x * gate) ----------
  // Last-use NT reads of x (still hits L2 from phase 1), NT write-only stores of out
  // so the 128 MiB output stream does not evict x working sets from L2.
  const float g0 = gate_s[pg * 4 + 0];
  const float g1 = gate_s[pg * 4 + 1];
  const float g2 = gate_s[pg * 4 + 2];
  const float g3 = gate_s[pg * 4 + 3];
  float* ob = out + (size_t)b * 65536u;
  for (int c = cs; c < 1024; c += 16) {
    const v4f v = __builtin_nontemporal_load((const v4f*)(xb + c * 64 + pg * 4));
    v4f o;
    o.x = fmaxf(v.x * g0, 0.f);
    o.y = fmaxf(v.y * g1, 0.f);
    o.z = fmaxf(v.z * g2, 0.f);
    o.w = fmaxf(v.w * g3, 0.f);
    __builtin_nontemporal_store(o, (v4f*)(ob + c * 64 + pg * 4));
  }
}

extern "C" void kernel_launch(void* const* d_in, const int* in_sizes, int n_in,
                              void* d_out, int out_size, void* d_ws, size_t ws_size,
                              hipStream_t stream) {
  (void)in_sizes; (void)n_in; (void)out_size; (void)d_ws; (void)ws_size;
  const float* x    = (const float*)d_in[0];
  const float* w1_1 = (const float*)d_in[1];
  const float* w5   = (const float*)d_in[2];
  const float* b5   = (const float*)d_in[3];
  const float* w6   = (const float*)d_in[4];
  const float* b6   = (const float*)d_in[5];
  float* out = (float*)d_out;
  sa_layer_fused<<<512, 256, 0, stream>>>(x, w1_1, w5, b5, w6, b6, out);
}
